// GraphSAGEFraudDetector_39676907880716
// MI455X (gfx1250) — compile-verified, matrix operand order checked by reading, not run on hardware
//
#include <hip/hip_runtime.h>

#define N_NODES 100000
#define N_EDGES 1000000
#define IN_CH   166
#define HID     128
#define OUTC    2
#define KP0     192   // IN_CH padded to multiple of 32
#define KP1     128

typedef __attribute__((ext_vector_type(16))) __bf16 v16bf;
typedef __attribute__((ext_vector_type(8)))  __bf16 v8bf;
typedef __attribute__((ext_vector_type(8)))  float  v8f;

union BF16x16 { v16bf v; v8bf h[2]; };

__global__ void zero_f32(float* p, int n) {
  for (int i = blockIdx.x * blockDim.x + threadIdx.x; i < n; i += gridDim.x * blockDim.x)
    p[i] = 0.f;
}

__global__ void deg_count(const int* __restrict__ dst, float* __restrict__ deg, int E) {
  for (int e = blockIdx.x * blockDim.x + threadIdx.x; e < E; e += gridDim.x * blockDim.x)
    atomicAdd(&deg[dst[e]], 1.0f);
}

__global__ void recip_deg(float* deg, int n) {
  for (int i = blockIdx.x * blockDim.x + threadIdx.x; i < n; i += gridDim.x * blockDim.x)
    deg[i] = 1.0f / fmaxf(deg[i], 1.0f);
}

// one block per edge; thread k adds feat[src][k] into agg[dst][k] (L2-resident atomics)
__global__ void scatter_add(const float* __restrict__ feat, const int* __restrict__ src,
                            const int* __restrict__ dst, float* __restrict__ agg,
                            int E, int C) {
  int e = blockIdx.x;
  int k = threadIdx.x;
  if (e >= E || k >= C) return;
  int s = src[e], d = dst[e];
  atomicAdd(&agg[(size_t)d * C + k], feat[(size_t)s * C + k]);
}

// build bf16 A matrices (row-major, K-padded): Am = mean = agg*rdeg, Ax = feat
__global__ void build_A(const float* __restrict__ feat, const float* __restrict__ agg,
                        const float* __restrict__ rdeg,
                        __bf16* __restrict__ Am, __bf16* __restrict__ Ax,
                        int C, int Kp) {
  int m = blockIdx.x;
  float rd = rdeg[m];
  for (int k = threadIdx.x; k < Kp; k += blockDim.x) {
    float xv = (k < C) ? feat[(size_t)m * C + k] : 0.f;
    float av = (k < C) ? agg[(size_t)m * C + k] * rd : 0.f;
    Ax[(size_t)m * Kp + k] = (__bf16)xv;
    Am[(size_t)m * Kp + k] = (__bf16)av;
  }
}

// pack fp32 weights [K x Nout] into WMMA B-fragment order:
// frag(kt,nt) = 32 lanes x 16 bf16; lane = N (mod 16), per-lane element j maps to
// K = kt*32 + (lane<16?0:8) + j + (j>=8?8:0)  (per CDNA5 16-bit B layout)
__global__ void pack_wfrag(const float* __restrict__ W, __bf16* __restrict__ F,
                           int K, int Nout, int ktiles, int ntiles) {
  int total = ktiles * ntiles * 512;
  for (int idx = blockIdx.x * blockDim.x + threadIdx.x; idx < total;
       idx += gridDim.x * blockDim.x) {
    int j    = idx & 15;
    int lane = (idx >> 4) & 31;
    int frag = idx >> 9;
    int nt = frag % ntiles;
    int kt = frag / ntiles;
    int n = nt * 16 + (lane & 15);
    int k = kt * 32 + ((lane < 16) ? 0 : 8) + j + ((j >= 8) ? 8 : 0);
    float v = (k < K && n < Nout) ? W[(size_t)k * Nout + n] : 0.f;
    F[idx] = (__bf16)v;
  }
}

// fused dual-GEMM: out = relu?( Am@Wl + Ax@Wr + bias )
// block = 256 threads (8 waves); wave wv owns N-tile wv; grid-stride over M-tiles.
// Two independent accumulator chains (mean-path, x-path) so consecutive WMMAs have
// no D->C serial dependency; summed in the epilogue.
__global__ __launch_bounds__(256) void sage_gemm(
    const __bf16* __restrict__ Am, const __bf16* __restrict__ Ax,
    const __bf16* __restrict__ WlF, const __bf16* __restrict__ WrF,
    const float* __restrict__ bias, float* __restrict__ out,
    int Kp, int Nout, int ntiles, int relu) {
  extern __shared__ __bf16 smem[];
  int ktiles = Kp >> 5;
  int fragElems = ktiles * ntiles * 512;
  {
    const unsigned* a = (const unsigned*)WlF;
    const unsigned* b = (const unsigned*)WrF;
    unsigned* s = (unsigned*)smem;
    int half = fragElems >> 1;
    for (int i = threadIdx.x; i < half; i += blockDim.x) {
      s[i] = a[i];
      s[half + i] = b[i];
    }
  }
  __syncthreads();

  int wv = threadIdx.x >> 5;
  int lane = threadIdx.x & 31;
  if (wv >= ntiles) return;                 // wave-uniform; EXEC stays full for WMMA

  int koff = (lane < 16) ? 0 : 8;           // A layout: lanes 0-15 K0-7/16-23; 16-31 K8-15/24-31
  int mtiles = N_NODES / 16;                // 6250 exactly
  for (int mt = blockIdx.x; mt < mtiles; mt += gridDim.x) {
    const __bf16* pm = Am + (size_t)(mt * 16 + (lane & 15)) * Kp;
    const __bf16* px = Ax + (size_t)(mt * 16 + (lane & 15)) * Kp;
    v8f accM = {0.f, 0.f, 0.f, 0.f, 0.f, 0.f, 0.f, 0.f};
    v8f accX = {0.f, 0.f, 0.f, 0.f, 0.f, 0.f, 0.f, 0.f};
    for (int kt = 0; kt < ktiles; ++kt) {
      int kb = kt * 32 + koff;
      BF16x16 am, ax;
      am.h[0] = *(const v8bf*)(pm + kb);
      am.h[1] = *(const v8bf*)(pm + kb + 16);
      ax.h[0] = *(const v8bf*)(px + kb);
      ax.h[1] = *(const v8bf*)(px + kb + 16);
      const __bf16* sl = smem + (size_t)(kt * ntiles + wv) * 512 + lane * 16;
      v16bf bl = *(const v16bf*)sl;
      v16bf br = *(const v16bf*)(sl + fragElems);
      accM = __builtin_amdgcn_wmma_f32_16x16x32_bf16(false, am.v, false, bl,
                                                     (short)0, accM, false, false);
      accX = __builtin_amdgcn_wmma_f32_16x16x32_bf16(false, ax.v, false, br,
                                                     (short)0, accX, false, false);
    }
    // C/D layout: VGPR v -> row v (lanes 0-15) / row 8+v (lanes 16-31); col = lane%16
    int col = wv * 16 + (lane & 15);
    float bv = (col < Nout) ? bias[col] : 0.f;
    int rbase = mt * 16 + ((lane < 16) ? 0 : 8);
#pragma unroll
    for (int v = 0; v < 8; ++v) {
      float val = accM[v] + accX[v] + bv;
      if (relu) val = fmaxf(val, 0.f);
      if (col < Nout) out[(size_t)(rbase + v) * Nout + col] = val;
    }
  }
}

extern "C" void kernel_launch(void* const* d_in, const int* in_sizes, int n_in,
                              void* d_out, int out_size, void* d_ws, size_t ws_size,
                              hipStream_t stream) {
  (void)in_sizes; (void)n_in; (void)out_size; (void)ws_size;
  const float* x   = (const float*)d_in[0];
  const int*   ei  = (const int*)d_in[1];
  const int*   src = ei;
  const int*   dst = ei + N_EDGES;
  const float* Wl0 = (const float*)d_in[2];
  const float* bl0 = (const float*)d_in[3];
  const float* Wr0 = (const float*)d_in[4];
  const float* Wl1 = (const float*)d_in[5];
  const float* bl1 = (const float*)d_in[6];
  const float* Wr1 = (const float*)d_in[7];
  const float* Wl2 = (const float*)d_in[8];
  const float* bl2 = (const float*)d_in[9];
  const float* Wr2 = (const float*)d_in[10];

  char* ws = (char*)d_ws;
  size_t off = 0;
  auto take = [&](size_t bytes) {
    char* p = ws + off;
    off = (off + bytes + 255) & ~(size_t)255;
    return p;
  };
  float*  rdeg = (float*)take((size_t)N_NODES * 4);
  float*  agg  = (float*)take((size_t)N_NODES * IN_CH * 4);   // reused all layers
  __bf16* Am   = (__bf16*)take((size_t)N_NODES * KP0 * 2);    // reused
  __bf16* Ax   = (__bf16*)take((size_t)N_NODES * KP0 * 2);    // reused
  float*  h0   = (float*)take((size_t)N_NODES * HID * 4);
  float*  h1   = (float*)take((size_t)N_NODES * HID * 4);
  const int kt0 = KP0 / 32, kt1 = KP1 / 32;
  __bf16* Fl0 = (__bf16*)take((size_t)kt0 * 8 * 512 * 2);
  __bf16* Fr0 = (__bf16*)take((size_t)kt0 * 8 * 512 * 2);
  __bf16* Fl1 = (__bf16*)take((size_t)kt1 * 8 * 512 * 2);
  __bf16* Fr1 = (__bf16*)take((size_t)kt1 * 8 * 512 * 2);
  __bf16* Fl2 = (__bf16*)take((size_t)kt1 * 1 * 512 * 2);
  __bf16* Fr2 = (__bf16*)take((size_t)kt1 * 1 * 512 * 2);

  // degrees -> rdeg = 1/max(deg,1)
  zero_f32<<<1024, 256, 0, stream>>>(rdeg, N_NODES);
  deg_count<<<4096, 256, 0, stream>>>(dst, rdeg, N_EDGES);
  recip_deg<<<1024, 256, 0, stream>>>(rdeg, N_NODES);

  // pack weights into WMMA fragment order (bf16)
  pack_wfrag<<<(kt0 * 8 * 512 + 255) / 256, 256, 0, stream>>>(Wl0, Fl0, IN_CH, HID, kt0, 8);
  pack_wfrag<<<(kt0 * 8 * 512 + 255) / 256, 256, 0, stream>>>(Wr0, Fr0, IN_CH, HID, kt0, 8);
  pack_wfrag<<<(kt1 * 8 * 512 + 255) / 256, 256, 0, stream>>>(Wl1, Fl1, HID, HID, kt1, 8);
  pack_wfrag<<<(kt1 * 8 * 512 + 255) / 256, 256, 0, stream>>>(Wr1, Fr1, HID, HID, kt1, 8);
  pack_wfrag<<<(kt1 * 1 * 512 + 255) / 256, 256, 0, stream>>>(Wl2, Fl2, HID, OUTC, kt1, 1);
  pack_wfrag<<<(kt1 * 1 * 512 + 255) / 256, 256, 0, stream>>>(Wr2, Fr2, HID, OUTC, kt1, 1);

  // ---- layer 0: x[166] -> h0[128], relu
  zero_f32<<<8192, 256, 0, stream>>>(agg, N_NODES * IN_CH);
  scatter_add<<<N_EDGES, 192, 0, stream>>>(x, src, dst, agg, N_EDGES, IN_CH);
  build_A<<<N_NODES, 192, 0, stream>>>(x, agg, rdeg, Am, Ax, IN_CH, KP0);
  sage_gemm<<<1024, 256, 2 * kt0 * 8 * 512 * 2, stream>>>(Am, Ax, Fl0, Fr0, bl0, h0,
                                                          KP0, HID, 8, 1);

  // ---- layer 1: h0[128] -> h1[128], relu
  zero_f32<<<8192, 256, 0, stream>>>(agg, N_NODES * HID);
  scatter_add<<<N_EDGES, 128, 0, stream>>>(h0, src, dst, agg, N_EDGES, HID);
  build_A<<<N_NODES, 128, 0, stream>>>(h0, agg, rdeg, Am, Ax, HID, KP1);
  sage_gemm<<<1024, 256, 2 * kt1 * 8 * 512 * 2, stream>>>(Am, Ax, Fl1, Fr1, bl1, h1,
                                                          KP1, HID, 8, 1);

  // ---- layer 2: h1[128] -> out[2] (logits)
  zero_f32<<<8192, 256, 0, stream>>>(agg, N_NODES * HID);
  scatter_add<<<N_EDGES, 128, 0, stream>>>(h1, src, dst, agg, N_EDGES, HID);
  build_A<<<N_NODES, 128, 0, stream>>>(h1, agg, rdeg, Am, Ax, HID, KP1);
  sage_gemm<<<1024, 256, 2 * kt1 * 1 * 512 * 2, stream>>>(Am, Ax, Fl2, Fr2, bl2,
                                                          (float*)d_out, KP1, OUTC, 1, 0);
}
